// ResNetMambaAgent_7430293422525
// MI455X (gfx1250) — matrix operand order repaired and checked by reading, not compiled, not run on hardware
//
#include <hip/hip_runtime.h>
#include <hip/hip_bf16.h>
#include <math.h>

// ---------------------------------------------------------------------------
// ResNet+Mamba agent forward for MI455X (gfx1250, wave32, WMMA).
// bf16 WMMA (v_wmma_f32_16x16x32_bf16) with fp32 accumulation for all convs
// (implicit GEMM, NHWC activations, (kh,kw,c)-ordered K so A-fragments are
// contiguous 16B vector loads) and dense GEMMs. 2-way M-blocking per wave
// shares B fragments across two accumulators. Norms/scan/elementwise fp32.
// ---------------------------------------------------------------------------

typedef __attribute__((ext_vector_type(16))) __bf16 v16bf;
typedef __attribute__((ext_vector_type(8)))  __bf16 v8bf;
typedef __attribute__((ext_vector_type(4)))  __bf16 v4bf;
typedef __attribute__((ext_vector_type(8)))  float  v8f;

#define BN_EPS  1e-5f
#define RMS_EPS 1e-6f

#define D_MODEL 256
#define D_INNER 512
#define D_STATE 128
#define HEADS   8
#define HEADDIM 64
#define DCONV   4
#define IN_DIM  1288   // 2*512 + 2*128 + 8
#define CONVDIM 768    // 512 + 2*128
#define TSEQ    256
#define BATCH   2
#define NIMG    512
#define CHUNK   64

__device__ __forceinline__ float silu_f(float x) { return x / (1.f + __expf(-x)); }

// ---------------------------------------------------------------------------
// Pack a row-major [N][K] fp32 matrix into the wave32 WMMA B-fragment layout:
// dst[((nt*nch + ch)*32 + lane)*16 + e]; element e of lane L is
// K = ch*32 + (L>=16 ? 8 : 0) + (e&7) + (e>=8 ? 16 : 0), N = nt*16 + (L&15). Zero-padded.
// ---------------------------------------------------------------------------
__global__ void pack_b_k(const float* __restrict__ w, __bf16* __restrict__ dst,
                         int N, int K) {
  int nch = (K + 31) >> 5;
  long total = (long)((N + 15) >> 4) * nch * 512;
  long idx = (long)blockIdx.x * blockDim.x + threadIdx.x;
  if (idx >= total) return;
  int e = idx & 15;
  int lane = (idx >> 4) & 31;
  long rest = idx >> 9;
  int ch = (int)(rest % nch);
  int nt = (int)(rest / nch);
  int k = ch * 32 + ((lane & 16) ? 8 : 0) + (e & 7) + ((e & 8) ? 16 : 0);
  int n = nt * 16 + (lane & 15);
  float v = (k < K && n < N) ? w[(size_t)n * K + k] : 0.f;
  dst[idx] = (__bf16)v;
}

// Conv weight pack with permuted K: k' = (kh*KS+kw)*Cpad + c  (c < Cin real).
// Weights stored [Cout][Cin][KS][KS].
__global__ void pack_conv_k(const float* __restrict__ w, __bf16* __restrict__ dst,
                            int Cout, int Cin, int Cpad, int KS) {
  int Kp = KS * KS * Cpad;
  int nch = (Kp + 31) >> 5;
  long total = (long)(Cout >> 4) * nch * 512;
  long idx = (long)blockIdx.x * blockDim.x + threadIdx.x;
  if (idx >= total) return;
  int e = idx & 15;
  int lane = (idx >> 4) & 31;
  long rest = idx >> 9;
  int ch = (int)(rest % nch);
  int nt = (int)(rest / nch);
  int kp = ch * 32 + ((lane & 16) ? 8 : 0) + (e & 7) + ((e & 8) ? 16 : 0);
  int n = nt * 16 + (lane & 15);
  float v = 0.f;
  if (kp < Kp) {
    int sp = kp / Cpad;
    int c = kp - sp * Cpad;
    if (c < Cin) {
      int kh = sp / KS, kw = sp - (sp / KS) * KS;
      v = w[(((size_t)n * Cin + c) * KS + kh) * KS + kw];
    }
  }
  dst[idx] = (__bf16)v;
}

// Fold BN into per-channel scale/bias: y = x*s + b
__global__ void bn_prep_k(const float* __restrict__ g, const float* __restrict__ b,
                          const float* __restrict__ m, const float* __restrict__ v,
                          float* __restrict__ sc, float* __restrict__ bi, int C) {
  int i = blockIdx.x * blockDim.x + threadIdx.x;
  if (i < C) {
    float s = g[i] * rsqrtf(v[i] + BN_EPS);
    sc[i] = s;
    bi[i] = b[i] - m[i] * s;
  }
}

// NCHW f32 -> NHWC bf16 (pad channels to 4) for the stem conv input
__global__ void nchw_to_nhwc4_k(const float* __restrict__ in, __bf16* __restrict__ out,
                                int nimg, int Cin, int H, int W) {
  long idx = (long)blockIdx.x * blockDim.x + threadIdx.x;
  long total = (long)nimg * H * W * 4;
  if (idx >= total) return;
  int c = (int)(idx & 3);
  long p = idx >> 2;
  int x = (int)(p % W); p /= W;
  int y = (int)(p % H);
  int img = (int)(p / H);
  float v = (c < Cin) ? in[(((size_t)img * Cin + c) * H + y) * W + x] : 0.f;
  out[idx] = (__bf16)v;
}

// ---------------------------------------------------------------------------
// A-fragment gather for the conv (NHWC, K order (kh,kw,c), Cpad = 1<<cs).
// ---------------------------------------------------------------------------
template <int CG>
__device__ __forceinline__ v16bf conv_gather(const __bf16* __restrict__ in,
                                             int img, int H, int W, int cs,
                                             int KS, int spmax, int stride, int pad,
                                             int oy, int ox, int kb) {
  const int Cpad = 1 << cs;
  v16bf a;
#pragma unroll
  for (int g = 0; g < 2; ++g) {
    const int ks = kb + g * 16;
#pragma unroll
    for (int s = 0; s < 8; s += CG) {
      const int kp = ks + s;
      const int sp = kp >> cs;
      const int c0 = kp & (Cpad - 1);
      const int kh = sp / KS, kw = sp - (sp / KS) * KS;
      const int iy = oy * stride - pad + kh;
      const int ix = ox * stride - pad + kw;
      const bool ok = (sp < spmax) &&
                      ((unsigned)iy < (unsigned)H) && ((unsigned)ix < (unsigned)W);
      if (CG == 8) {
        v8bf av = {};
        if (ok) av = *(const v8bf*)(in + ((((size_t)img * H + iy) * W + ix) << cs) + c0);
#pragma unroll
        for (int j = 0; j < 8; ++j) a[g * 8 + s + j] = av[j];
      } else {
        v4bf av = {};
        if (ok) av = *(const v4bf*)(in + ((((size_t)img * H + iy) * W + ix) << cs) + c0);
#pragma unroll
        for (int j = 0; j < 4; ++j) a[g * 8 + s + j] = av[j];
      }
    }
  }
  return a;
}

// ---------------------------------------------------------------------------
// Implicit-GEMM conv via WMMA, NHWC activations. One wave computes TWO
// 16(pixel) x 16(channel) tiles stacked in M, sharing B fragments.
// All tile dims (Ho*Wo, Cout) are multiples of 16 by construction.
// ---------------------------------------------------------------------------
template <int CG>
__global__ void conv_wmma_k(const __bf16* __restrict__ in,
                            const __bf16* __restrict__ wpack,
                            const float* __restrict__ scale,
                            const float* __restrict__ bias,
                            const __bf16* __restrict__ res,
                            __bf16* __restrict__ out,
                            int cs,        // log2(Cpad)
                            int Cout, int H, int W, int Ho, int Wo,
                            int KS, int stride, int pad, int relu) {
  const int lane = threadIdx.x;
  const int nt = blockIdx.y, img = blockIdx.z;
  const int mN = Ho * Wo;
  const int mt0 = blockIdx.x * 2;
  const bool has2 = (mt0 + 1) < (mN >> 4);   // uniform across the wave
  const int m0 = mt0 * 16 + (lane & 15);
  const int m1 = m0 + 16;
  const int oy0 = m0 / Wo, ox0 = m0 - (m0 / Wo) * Wo;
  const int oy1 = m1 / Wo, ox1 = m1 - (m1 / Wo) * Wo;
  const int spmax = KS * KS;
  const int Kreal = spmax << cs;
  const int nchunks = (Kreal + 31) >> 5;

  v8f acc0 = {}, acc1 = {};
  const __bf16* bp = wpack + (((size_t)nt * nchunks) * 32 + lane) * 16;
  for (int ch = 0; ch < nchunks; ++ch) {
    v16bf b = *(const v16bf*)bp;
    bp += 512;
    if (ch + 1 < nchunks) __builtin_prefetch(bp, 0, 1);
    const int kb = ch * 32 + ((lane & 16) ? 8 : 0);
    v16bf a0 = conv_gather<CG>(in, img, H, W, cs, KS, spmax, stride, pad, oy0, ox0, kb);
    acc0 = __builtin_amdgcn_wmma_f32_16x16x32_bf16(false, a0, false, b,
                                                   (short)0, acc0, false, false);
    if (has2) {
      v16bf a1 = conv_gather<CG>(in, img, H, W, cs, KS, spmax, stride, pad, oy1, ox1, kb);
      acc1 = __builtin_amdgcn_wmma_f32_16x16x32_bf16(false, a1, false, b,
                                                     (short)0, acc1, false, false);
    }
  }

  const int n = nt * 16 + (lane & 15);
  const float sc = scale[n], bi = bias[n];
  const int mr0 = mt0 * 16 + ((lane & 16) ? 8 : 0);
#pragma unroll
  for (int r = 0; r < 8; ++r) {
    const size_t o = ((size_t)img * mN + (mr0 + r)) * Cout + n;   // NHWC
    float v = acc0[r] * sc + bi;
    if (res) v += (float)res[o];
    if (relu) v = fmaxf(v, 0.f);
    out[o] = (__bf16)v;
  }
  if (has2) {
#pragma unroll
    for (int r = 0; r < 8; ++r) {
      const size_t o = ((size_t)img * mN + (mr0 + 16 + r)) * Cout + n;
      float v = acc1[r] * sc + bi;
      if (res) v += (float)res[o];
      if (relu) v = fmaxf(v, 0.f);
      out[o] = (__bf16)v;
    }
  }
}

// ---------------------------------------------------------------------------
// Dense GEMM via WMMA: out[M][N] = A[M][K] @ W[N][K]^T (+bias)(+res).
// A is bf16 row-major; M % 32 == 0 and K % 32 == 0 guaranteed by callers.
// 2-way M-blocking per wave: B fragment shared across two accumulators.
// ---------------------------------------------------------------------------
__global__ void gemm_wmma_k(const __bf16* __restrict__ A,
                            const __bf16* __restrict__ Bp,
                            const float* __restrict__ bias,
                            const float* __restrict__ res,
                            float* __restrict__ out, int M, int N, int K) {
  const int lane = threadIdx.x;
  const int nt = blockIdx.y;
  const int mt0 = blockIdx.x * 2;
  const int m0 = mt0 * 16 + (lane & 15);
  const int nch = K >> 5;

  v8f acc0 = {}, acc1 = {};
  const __bf16* bp = Bp + (((size_t)nt * nch) * 32 + lane) * 16;
  const __bf16* arow0 = A + (size_t)m0 * K + ((lane & 16) ? 8 : 0);
  const __bf16* arow1 = arow0 + (size_t)16 * K;
  for (int ch = 0; ch < nch; ++ch) {
    v16bf b = *(const v16bf*)bp;
    bp += 512;
    if (ch + 1 < nch) __builtin_prefetch(bp, 0, 1);
    v8bf lo0 = *(const v8bf*)(arow0);
    v8bf hi0 = *(const v8bf*)(arow0 + 16);
    v8bf lo1 = *(const v8bf*)(arow1);
    v8bf hi1 = *(const v8bf*)(arow1 + 16);
    arow0 += 32;
    arow1 += 32;
    v16bf a0 = __builtin_shufflevector(lo0, hi0, 0, 1, 2, 3, 4, 5, 6, 7,
                                       8, 9, 10, 11, 12, 13, 14, 15);
    v16bf a1 = __builtin_shufflevector(lo1, hi1, 0, 1, 2, 3, 4, 5, 6, 7,
                                       8, 9, 10, 11, 12, 13, 14, 15);
    acc0 = __builtin_amdgcn_wmma_f32_16x16x32_bf16(false, a0, false, b,
                                                   (short)0, acc0, false, false);
    acc1 = __builtin_amdgcn_wmma_f32_16x16x32_bf16(false, a1, false, b,
                                                   (short)0, acc1, false, false);
  }

  const int n = nt * 16 + (lane & 15);
  if (n >= N) return;
  const int mr0 = mt0 * 16 + ((lane & 16) ? 8 : 0);
  const float bb = bias ? bias[n] : 0.f;
#pragma unroll
  for (int r = 0; r < 8; ++r) {
    const int mm = mr0 + r;
    float v = acc0[r] + bb;
    if (res) v += res[(size_t)mm * N + n];
    out[(size_t)mm * N + n] = v;
  }
#pragma unroll
  for (int r = 0; r < 8; ++r) {
    const int mm = mr0 + 16 + r;
    float v = acc1[r] + bb;
    if (res) v += res[(size_t)mm * N + n];
    out[(size_t)mm * N + n] = v;
  }
}

// 3x3 stride-2 pad-1 max pool, NHWC bf16
__global__ void maxpool_k(const __bf16* __restrict__ in, __bf16* __restrict__ out,
                          int nimg, int C, int H, int W, int Ho, int Wo) {
  long idx = (long)blockIdx.x * blockDim.x + threadIdx.x;
  long total = (long)nimg * C * Ho * Wo;
  if (idx >= total) return;
  int c = (int)(idx % C);
  long t = idx / C;
  int wo = (int)(t % Wo); t /= Wo;
  int ho = (int)(t % Ho);
  int img = (int)(t / Ho);
  float mx = -3.4e38f;
#pragma unroll
  for (int kh = 0; kh < 3; ++kh)
#pragma unroll
    for (int kw = 0; kw < 3; ++kw) {
      int iy = ho * 2 - 1 + kh, ix = wo * 2 - 1 + kw;
      if ((unsigned)iy < (unsigned)H && (unsigned)ix < (unsigned)W)
        mx = fmaxf(mx, (float)in[(((size_t)img * H + iy) * W + ix) * C + c]);
    }
  out[idx] = (__bf16)mx;   // out is [img][ho][wo][c] == idx
}

// Global average pool NHWC: [nimg][HW][C] bf16 -> bf16 feat rows at g0
__global__ void avgpool_k(const __bf16* __restrict__ in, __bf16* __restrict__ feat,
                          int nimg, int C, int HW, int g0) {
  int idx = blockIdx.x * blockDim.x + threadIdx.x;
  if (idx >= nimg * C) return;
  int c = idx % C, i = idx / C;
  float s = 0.f;
  for (int j = 0; j < HW; ++j) s += (float)in[((size_t)i * HW + j) * C + c];
  feat[(size_t)(g0 + i) * C + c] = (__bf16)(s / (float)HW);
}

// RMSNorm over 256 dims (one block per row), f32 in -> bf16 out
__global__ void rmsnorm256_k(const float* __restrict__ x, const float* __restrict__ w,
                             __bf16* __restrict__ out, int rows) {
  int row = blockIdx.x;
  int c = threadIdx.x;
  __shared__ float red[256];
  float v = x[(size_t)row * 256 + c];
  red[c] = v * v;
  __syncthreads();
  for (int s = 128; s > 0; s >>= 1) {
    if (c < s) red[c] += red[c + s];
    __syncthreads();
  }
  float inv = rsqrtf(red[0] / 256.f + RMS_EPS);
  out[(size_t)row * 256 + c] = (__bf16)(v * inv * w[c]);
}

// Causal depthwise conv1d (width 4) + SiLU over xBC slice of zxbcdt
__global__ void convdw_silu_k(const float* __restrict__ zx, const float* __restrict__ w,
                              const float* __restrict__ b, float* __restrict__ out) {
  long idx = (long)blockIdx.x * blockDim.x + threadIdx.x;
  if (idx >= (long)NIMG * CONVDIM) return;
  int c = (int)(idx % CONVDIM);
  int row = (int)(idx / CONVDIM);
  int t = row & (TSEQ - 1);
  float y = b[c];
#pragma unroll
  for (int j = 0; j < DCONV; ++j) {
    int tt = t - (DCONV - 1) + j;
    if (tt >= 0)
      y += w[c * DCONV + j] * zx[(size_t)(row - (DCONV - 1) + j) * IN_DIM + D_INNER + c];
  }
  out[idx] = silu_f(y);
}

// Sequential SSM scan: one workgroup per (batch, head); state in registers.
__global__ void ssm_scan_k(const float* __restrict__ zx, const float* __restrict__ xc,
                           const float* __restrict__ dt_bias, const float* __restrict__ A_log,
                           const float* __restrict__ Dp, float* __restrict__ yraw) {
  int bh = blockIdx.x;
  int b = bh >> 3, h = bh & 7;
  int tid = threadIdx.x;
  int p = tid >> 2, nq = tid & 3;
  __shared__ float Bs[D_STATE], Cs[D_STATE], xs[HEADDIM], red[256];
  float hreg[32];
#pragma unroll
  for (int j = 0; j < 32; ++j) hreg[j] = 0.f;
  const float nA = -__expf(A_log[h]);
  const float db = dt_bias[h];
  const float Dh = Dp[h];
  const int n0 = nq * 32;
  for (int t = 0; t < TSEQ; ++t) {
    int row = b * TSEQ + t;
    if (tid < 128) Bs[tid] = xc[(size_t)row * CONVDIM + D_INNER + tid];
    else           Cs[tid - 128] = xc[(size_t)row * CONVDIM + D_INNER + D_STATE + (tid - 128)];
    if (tid < 64)  xs[tid] = xc[(size_t)row * CONVDIM + h * HEADDIM + tid];
    __syncthreads();
    float dtr = zx[(size_t)row * IN_DIM + 2 * D_INNER + 2 * D_STATE + h] + db;
    float dt = (dtr > 20.f) ? dtr : log1pf(__expf(dtr));
    float dA = __expf(dt * nA);
    float coeff = dt * xs[p];
    float part = 0.f;
#pragma unroll
    for (int j = 0; j < 32; ++j) {
      hreg[j] = hreg[j] * dA + coeff * Bs[n0 + j];
      part += hreg[j] * Cs[n0 + j];
    }
    red[tid] = part;
    __syncthreads();
    if (nq == 0) {
      float y = red[tid] + red[tid + 1] + red[tid + 2] + red[tid + 3] + Dh * xs[p];
      yraw[(size_t)row * D_INNER + h * HEADDIM + p] = y;
    }
    __syncthreads();
  }
}

// y * silu(z), then RMSNorm over 512 dims with mixer_norm_w -> bf16
__global__ void gate_norm_k(const float* __restrict__ yraw, const float* __restrict__ zx,
                            const float* __restrict__ w, __bf16* __restrict__ out) {
  int row = blockIdx.x;
  int c = threadIdx.x;
  __shared__ float red[256];
  float z1 = zx[(size_t)row * IN_DIM + c];
  float z2 = zx[(size_t)row * IN_DIM + c + 256];
  float g1 = yraw[(size_t)row * D_INNER + c] * silu_f(z1);
  float g2 = yraw[(size_t)row * D_INNER + c + 256] * silu_f(z2);
  red[c] = g1 * g1 + g2 * g2;
  __syncthreads();
  for (int s = 128; s > 0; s >>= 1) {
    if (c < s) red[c] += red[c + s];
    __syncthreads();
  }
  float inv = rsqrtf(red[0] / (float)D_INNER + RMS_EPS);
  out[(size_t)row * D_INNER + c] = (__bf16)(g1 * inv * w[c]);
  out[(size_t)row * D_INNER + c + 256] = (__bf16)(g2 * inv * w[c + 256]);
}

// Final RMSNorm of the last token + actor/critic heads. out: logits(2,4)+value(2)
__global__ void head_k(const float* __restrict__ x, const float* __restrict__ norm_f,
                       const float* __restrict__ aw, const float* __restrict__ ab,
                       const float* __restrict__ cw, const float* __restrict__ cb,
                       float* __restrict__ out) {
  int b = blockIdx.x;
  int c = threadIdx.x;
  __shared__ float red[256];
  __shared__ float xn[256];
  int row = b * TSEQ + (TSEQ - 1);
  float v = x[(size_t)row * D_MODEL + c];
  red[c] = v * v;
  __syncthreads();
  for (int s = 128; s > 0; s >>= 1) {
    if (c < s) red[c] += red[c + s];
    __syncthreads();
  }
  float inv = rsqrtf(red[0] / (float)D_MODEL + RMS_EPS);
  xn[c] = v * inv * norm_f[c];
  __syncthreads();
  for (int o = 0; o < 5; ++o) {
    red[c] = xn[c] * (o < 4 ? aw[o * 256 + c] : cw[c]);
    __syncthreads();
    for (int s = 128; s > 0; s >>= 1) {
      if (c < s) red[c] += red[c + s];
      __syncthreads();
    }
    if (c == 0) {
      if (o < 4) out[b * 4 + o] = red[0] + ab[o];
      else       out[8 + b] = red[0] + cb[0];
    }
    __syncthreads();
  }
}

// ---------------------------------------------------------------------------
// Host side
// ---------------------------------------------------------------------------
struct BNp { const float *g, *b, *m, *v; };
struct Blk { const float* c1; BNp b1; const float* c2; BNp b2; int sc; const float* scw; BNp sbn; };
struct ML  { const float *norm_w, *in_proj, *conv_w, *conv_b, *dt_bias, *A_log, *D, *mixer_norm_w, *out_proj; };

static inline size_t packElems(int N, int K) {
  return (size_t)((N + 15) / 16) * ((K + 31) / 32) * 512;
}

extern "C" void kernel_launch(void* const* d_in, const int* in_sizes, int n_in,
                              void* d_out, int out_size, void* d_ws, size_t ws_size,
                              hipStream_t stream) {
  (void)in_sizes; (void)n_in; (void)out_size; (void)ws_size;

  // ---- walk inputs in setup_inputs() dict insertion order ----
  int pi = 0;
  auto nxt = [&]() -> const float* { return (const float*)d_in[pi++]; };
  const float* img = nxt();
  const float* init_conv = nxt();
  BNp init_bn = {nxt(), nxt(), nxt(), nxt()};
  Blk blks[8];
  for (int li = 0; li < 4; ++li)
    for (int bi = 0; bi < 2; ++bi) {
      Blk& B = blks[li * 2 + bi];
      B.c1 = nxt(); B.b1 = {nxt(), nxt(), nxt(), nxt()};
      B.c2 = nxt(); B.b2 = {nxt(), nxt(), nxt(), nxt()};
      B.sc = (li > 0 && bi == 0);
      if (B.sc) { B.scw = nxt(); B.sbn = {nxt(), nxt(), nxt(), nxt()}; }
      else      { B.scw = nullptr; B.sbn = {nullptr, nullptr, nullptr, nullptr}; }
    }
  const float* fc_w = nxt();
  const float* fc_b = nxt();
  ML ml[6];
  for (int l = 0; l < 6; ++l) {
    ml[l].norm_w = nxt(); ml[l].in_proj = nxt(); ml[l].conv_w = nxt(); ml[l].conv_b = nxt();
    ml[l].dt_bias = nxt(); ml[l].A_log = nxt(); ml[l].D = nxt();
    ml[l].mixer_norm_w = nxt(); ml[l].out_proj = nxt();
  }
  const float* norm_f = nxt();
  const float* actor_w = nxt(); const float* actor_b = nxt();
  const float* critic_w = nxt(); const float* critic_b = nxt();

  // ---- carve workspace ----
  char* wp = (char*)d_ws;
  auto carve = [&](size_t bytes) -> void* {
    void* r = (void*)wp;
    wp += (bytes + 255) & ~(size_t)255;
    return r;
  };
  const int cins[4] = {32, 32, 64, 128};
  const int couts[4] = {32, 64, 128, 256};

  __bf16* bufIn = (__bf16*)carve(sizeof(__bf16) * (size_t)CHUNK * 128 * 128 * 4); // stem NHWC4
  __bf16* bufA  = (__bf16*)carve(sizeof(__bf16) * (size_t)CHUNK * 64 * 64 * 32);  // stem out / t1 temp
  __bf16* buf0  = (__bf16*)carve(sizeof(__bf16) * (size_t)CHUNK * 32 * 32 * 32);  // persistent "cur"
  __bf16* buf1  = (__bf16*)carve(sizeof(__bf16) * (size_t)CHUNK * 32 * 32 * 32);  // shortcut temp

  __bf16* pk_init = (__bf16*)carve(sizeof(__bf16) * packElems(32, 49 * 4));
  __bf16 *pk_c1[8], *pk_c2[8], *pk_sc[8];
  float *s1[8], *bb1[8], *s2[8], *bb2[8], *ssc[8], *bsc[8];
  for (int li = 0; li < 4; ++li)
    for (int bi = 0; bi < 2; ++bi) {
      int i = li * 2 + bi;
      int cin = (bi == 0) ? cins[li] : couts[li];
      int cout = couts[li];
      pk_c1[i] = (__bf16*)carve(sizeof(__bf16) * packElems(cout, cin * 9));
      pk_c2[i] = (__bf16*)carve(sizeof(__bf16) * packElems(cout, cout * 9));
      s1[i] = (float*)carve(4 * cout); bb1[i] = (float*)carve(4 * cout);
      s2[i] = (float*)carve(4 * cout); bb2[i] = (float*)carve(4 * cout);
      if (blks[i].sc) {
        pk_sc[i] = (__bf16*)carve(sizeof(__bf16) * packElems(cout, cin));
        ssc[i] = (float*)carve(4 * cout); bsc[i] = (float*)carve(4 * cout);
      } else { pk_sc[i] = nullptr; ssc[i] = nullptr; bsc[i] = nullptr; }
    }
  float* sinit = (float*)carve(4 * 32);
  float* binit = (float*)carve(4 * 32);
  __bf16* pk_fc = (__bf16*)carve(sizeof(__bf16) * packElems(256, 256));
  __bf16 *pk_in[6], *pk_op[6];
  for (int l = 0; l < 6; ++l) {
    pk_in[l] = (__bf16*)carve(sizeof(__bf16) * packElems(IN_DIM, 256));
    pk_op[l] = (__bf16*)carve(sizeof(__bf16) * packElems(256, 512));
  }
  __bf16* feat  = (__bf16*)carve(sizeof(__bf16) * (size_t)NIMG * 256);
  __bf16* xnrm  = (__bf16*)carve(sizeof(__bf16) * (size_t)NIMG * 256);
  __bf16* gn    = (__bf16*)carve(sizeof(__bf16) * (size_t)NIMG * 512);
  float* xbuf  = (float*)carve(4 * (size_t)NIMG * 256);
  float* xbuf2 = (float*)carve(4 * (size_t)NIMG * 256);
  float* zx    = (float*)carve(4 * (size_t)NIMG * IN_DIM);
  float* xc    = (float*)carve(4 * (size_t)NIMG * CONVDIM);
  float* yraw  = (float*)carve(4 * (size_t)NIMG * 512);

  // ---- launch helpers ----
  auto packD = [&](const float* w, __bf16* dst, int N, int K) {
    size_t tot = packElems(N, K);
    pack_b_k<<<(unsigned)((tot + 255) / 256), 256, 0, stream>>>(w, dst, N, K);
  };
  auto packC = [&](const float* w, __bf16* dst, int Cout, int Cin, int Cpad, int KS) {
    size_t tot = packElems(Cout, KS * KS * Cpad);
    pack_conv_k<<<(unsigned)((tot + 255) / 256), 256, 0, stream>>>(w, dst, Cout, Cin, Cpad, KS);
  };
  auto bnp = [&](const BNp& p, float* s, float* b, int C) {
    bn_prep_k<<<(C + 255) / 256, 256, 0, stream>>>(p.g, p.b, p.m, p.v, s, b, C);
  };
  auto conv = [&](const __bf16* in, int Cpad, const __bf16* wpk,
                  const float* s, const float* b, const __bf16* res, __bf16* out,
                  int Cout, int H, int Ho, int KS, int st, int pad, bool relu) {
    int mtiles = (Ho * Ho) / 16;
    dim3 g((mtiles + 1) / 2, Cout / 16, CHUNK);
    int cs = __builtin_ctz(Cpad);
    if (Cpad >= 8)
      conv_wmma_k<8><<<g, 32, 0, stream>>>(in, wpk, s, b, res, out, cs,
                                           Cout, H, H, Ho, Ho, KS, st, pad, relu ? 1 : 0);
    else
      conv_wmma_k<4><<<g, 32, 0, stream>>>(in, wpk, s, b, res, out, cs,
                                           Cout, H, H, Ho, Ho, KS, st, pad, relu ? 1 : 0);
  };
  auto gemm = [&](const __bf16* A, const __bf16* Bp, const float* bias,
                  const float* res, float* out, int M, int N, int K) {
    dim3 g(M / 32, (N + 15) / 16);
    gemm_wmma_k<<<g, 32, 0, stream>>>(A, Bp, bias, res, out, M, N, K);
  };

  // ---- weight packing + BN folding (once per call) ----
  packC(init_conv, pk_init, 32, 3, 4, 7);
  bnp(init_bn, sinit, binit, 32);
  for (int li = 0; li < 4; ++li)
    for (int bi = 0; bi < 2; ++bi) {
      int i = li * 2 + bi;
      int cin = (bi == 0) ? cins[li] : couts[li];
      int cout = couts[li];
      packC(blks[i].c1, pk_c1[i], cout, cin, cin, 3);
      packC(blks[i].c2, pk_c2[i], cout, cout, cout, 3);
      bnp(blks[i].b1, s1[i], bb1[i], cout);
      bnp(blks[i].b2, s2[i], bb2[i], cout);
      if (blks[i].sc) {
        packC(blks[i].scw, pk_sc[i], cout, cin, cin, 1);
        bnp(blks[i].sbn, ssc[i], bsc[i], cout);
      }
    }
  packD(fc_w, pk_fc, 256, 256);
  for (int l = 0; l < 6; ++l) {
    packD(ml[l].in_proj, pk_in[l], IN_DIM, 256);
    packD(ml[l].out_proj, pk_op[l], 256, 512);
  }

  // ---- ResNet over image chunks (NHWC activations) ----
  for (int g0 = 0; g0 < NIMG; g0 += CHUNK) {
    const float* inimg = img + (size_t)g0 * 3 * 128 * 128;
    {
      long tot = (long)CHUNK * 128 * 128 * 4;
      nchw_to_nhwc4_k<<<(unsigned)((tot + 255) / 256), 256, 0, stream>>>(inimg, bufIn, CHUNK, 3, 128, 128);
    }
    // stem conv 7x7 s2 p3 + BN + ReLU -> bufA (CHUNK x 64 x 64 x 32, NHWC)
    conv(bufIn, 4, pk_init, sinit, binit, nullptr, bufA, 32, 128, 64, 7, 2, 3, true);
    // maxpool 3x3 s2 p1 -> buf0 (CHUNK x 32 x 32 x 32)
    {
      long tot = (long)CHUNK * 32 * 32 * 32;
      maxpool_k<<<(unsigned)((tot + 255) / 256), 256, 0, stream>>>(bufA, buf0, CHUNK, 32, 64, 64, 32, 32);
    }
    int curH = 32;
    for (int li = 0; li < 4; ++li)
      for (int bi = 0; bi < 2; ++bi) {
        int i = li * 2 + bi;
        int cin = (bi == 0) ? cins[li] : couts[li];
        int cout = couts[li];
        int stride = (bi == 0 && li > 0) ? 2 : 1;
        int Hout = (stride == 2) ? curH / 2 : curH;
        const __bf16* idt = buf0;
        if (blks[i].sc) {
          conv(buf0, cin, pk_sc[i], ssc[i], bsc[i], nullptr, buf1,
               cout, curH, Hout, 1, 2, 0, false);
          idt = buf1;
        }
        conv(buf0, cin, pk_c1[i], s1[i], bb1[i], nullptr, bufA,
             cout, curH, Hout, 3, stride, 1, true);
        conv(bufA, cout, pk_c2[i], s2[i], bb2[i], idt, buf0,
             cout, Hout, Hout, 3, 1, 1, true);
        curH = Hout;
      }
    // global average pool (4x4 spatial, 256 ch) -> feat rows [g0 .. g0+CHUNK)
    avgpool_k<<<(CHUNK * 256 + 255) / 256, 256, 0, stream>>>(buf0, feat, CHUNK, 256, 16, g0);
  }

  // fc: x = feat @ fc_w^T + fc_b   (512 x 256)
  gemm(feat, pk_fc, fc_b, nullptr, xbuf, NIMG, 256, 256);

  // ---- Mamba stack ----
  for (int l = 0; l < 6; ++l) {
    rmsnorm256_k<<<NIMG, 256, 0, stream>>>(xbuf, ml[l].norm_w, xnrm, NIMG);
    gemm(xnrm, pk_in[l], nullptr, nullptr, zx, NIMG, IN_DIM, 256);
    {
      long tot = (long)NIMG * CONVDIM;
      convdw_silu_k<<<(unsigned)((tot + 255) / 256), 256, 0, stream>>>(zx, ml[l].conv_w, ml[l].conv_b, xc);
    }
    ssm_scan_k<<<BATCH * HEADS, 256, 0, stream>>>(zx, xc, ml[l].dt_bias, ml[l].A_log, ml[l].D, yraw);
    gate_norm_k<<<NIMG, 256, 0, stream>>>(yraw, zx, ml[l].mixer_norm_w, gn);
    gemm(gn, pk_op[l], nullptr, xbuf, xbuf2, NIMG, 256, 512);
    float* t = xbuf; xbuf = xbuf2; xbuf2 = t;
  }

  // ---- final norm + heads ----
  head_k<<<BATCH, 256, 0, stream>>>(xbuf, norm_f, actor_w, actor_b,
                                    critic_w, critic_b, (float*)d_out);
}
